// E_GCL_vel_44865228374415
// MI455X (gfx1250) — matrix operand order repaired and checked by reading, not compiled, not run on hardware
//
#include <hip/hip_runtime.h>
#include <stdint.h>

// Problem constants (from reference)
constexpr int Nn  = 20000;      // nodes
constexpr int Ne  = 320000;     // edges
constexpr int Bb  = 16;         // batches
constexpr int Cc  = 4;          // virtual channels
constexpr int Mv  = Nn * Cc;    // 80000 virtual rows
// NF == H == 128 throughout

typedef _Float16 v16h __attribute__((ext_vector_type(16)));
typedef _Float16 v8h  __attribute__((ext_vector_type(8)));
typedef _Float16 v4h  __attribute__((ext_vector_type(4)));
typedef float    v8f  __attribute__((ext_vector_type(8)));

__device__ __forceinline__ float siluf(float x){ return x * (1.0f/(1.0f + __expf(-x))); }
__device__ __forceinline__ float sigmf(float x){ return 1.0f/(1.0f + __expf(-x)); }

// Swizzled A layout: 16-row x 32-k tiles, each tile = 512 contiguous halves laid out
// exactly as the v_wmma 16-bit A fragment: [half(kk/16)][lane(m+16*((kk%16)/8))][j(kk%8)].
// KC = KPAD/32 chunks per row-tile.
__device__ __forceinline__ long aswz(int row, int k, int KC){
  int tile = row >> 4, m = row & 15;
  int chunk = k >> 5, kk = k & 31;
  return (((long)(tile * KC + chunk)) << 9)
       + ((kk >> 4) << 8) + ((m + ((kk >> 3) & 1) * 16) << 3) + (kk & 7);
}

// ---------------------------------------------------------------- zero
__global__ void k_zero(float* __restrict__ p, long n){
  long i = (long)blockIdx.x*blockDim.x + threadIdx.x;
  if (i < n) p[i] = 0.0f;
}

// ---------------------------------------------------------------- weight prep
// Wt[n][k] = (k<K) ? W[k][n] : 0   (f16, transposed, zero-padded to KPAD)
__global__ void k_wt(const float* __restrict__ W, _Float16* __restrict__ Wt, int K, int KPAD){
  long g = (long)blockIdx.x*blockDim.x + threadIdx.x;
  if (g >= (long)128*KPAD) return;
  int n = (int)(g / KPAD), k = (int)(g % KPAD);
  Wt[g] = (k < K) ? (_Float16)W[(long)k*128 + n] : (_Float16)0.0f;
}

// ---------------------------------------------------------------- WMMA GEMM
// out[M,128] = epilogue(A[M,KPAD](f16, swizzled) @ Wt^T + bias)
// Block = 256 threads = 8 waves; block tile = 256 rows x 128 cols (2 row-tiles/wave).
// Weight chunks (8KB f16) staged to LDS with global_load_async_to_lds_b128.
template<int KPAD, bool SILU>
__global__ __launch_bounds__(256) void gemm128(
    const _Float16* __restrict__ A,    // swizzled, KC = KPAD/32
    const _Float16* __restrict__ Wt,   // [128][KPAD] f16
    const float*    __restrict__ bias,
    const float*    __restrict__ resid,
    float*          __restrict__ outF, // row-major f32 or null
    _Float16*       __restrict__ outH, // swizzled (KC=4) f16 or null
    int M)
{
  constexpr int KC = KPAD / 32;
  __shared__ _Float16 Ws[128 * 32];     // transposed chunk: Ws[n*32 + klocal]
  const int tid  = threadIdx.x;
  const int wave = tid >> 5;
  const int lane = tid & 31;
  const int l16  = lane & 15;
  const int hi   = lane >> 4;
  const int r0   = blockIdx.x * 256 + wave * 32;   // first row-tile base
  const int r1   = r0 + 16;                        // second row-tile base

  const long t0 = (r0 < M) ? (long)(r0 >> 4) : 0;  // clamp; results discarded
  const long t1 = (r1 < M) ? (long)(r1 >> 4) : 0;
  const _Float16* a0 = A + (t0 * KC << 9) + lane * 8;
  const _Float16* a1 = A + (t1 * KC << 9) + lane * 8;

  // async W staging: thread -> (n, 16-half group)
  const int sn = tid >> 1, sh = tid & 1;
  const _Float16* wsrc = Wt + (size_t)sn * KPAD + sh * 16;
  unsigned lds0 = (unsigned)(uintptr_t)&Ws[sn * 32 + sh * 16];

  v8f acc0[8], acc1[8];
#pragma unroll
  for (int nt = 0; nt < 8; ++nt)
#pragma unroll
    for (int j = 0; j < 8; ++j){ acc0[nt][j] = 0.0f; acc1[nt][j] = 0.0f; }

  for (int c = 0; c < KC; ++c) {
    // stage 8KB weight chunk: global f16 -> LDS (async, 2x b128 per thread)
    {
      const _Float16* g = wsrc + c * 32;
      asm volatile("global_load_async_to_lds_b128 %0, %1, off"
                   :: "v"(lds0), "v"((unsigned long long)(uintptr_t)g) : "memory");
      asm volatile("global_load_async_to_lds_b128 %0, %1, off"
                   :: "v"(lds0 + 16u), "v"((unsigned long long)(uintptr_t)(g + 8)) : "memory");
      asm volatile("s_wait_asynccnt 0x0" ::: "memory");
    }
    __syncthreads();

    // A fragments: fully coalesced (each half-block is 512B contiguous per wave)
    const _Float16* ap0 = a0 + ((long)c << 9);
    const _Float16* ap1 = a1 + ((long)c << 9);
    v8h a0lo = *(const v8h*)(ap0);
    v8h a0hi = *(const v8h*)(ap0 + 256);
    v8h a1lo = *(const v8h*)(ap1);
    v8h a1hi = *(const v8h*)(ap1 + 256);
    v16h af0 = __builtin_shufflevector(a0lo, a0hi, 0,1,2,3,4,5,6,7,8,9,10,11,12,13,14,15);
    v16h af1 = __builtin_shufflevector(a1lo, a1hi, 0,1,2,3,4,5,6,7,8,9,10,11,12,13,14,15);
    if (c + 1 < KC) {
      __builtin_prefetch(a0 + ((long)(c + 1) << 9), 0, 0);
      __builtin_prefetch(a1 + ((long)(c + 1) << 9), 0, 0);
    }

#pragma unroll
    for (int nt = 0; nt < 8; ++nt) {
      const _Float16* bp = &Ws[(nt * 16 + l16) * 32 + hi * 8];
      v8h blo = *(const v8h*)(bp);
      v8h bhi = *(const v8h*)(bp + 16);
      v16h bfrag = __builtin_shufflevector(blo, bhi, 0,1,2,3,4,5,6,7,8,9,10,11,12,13,14,15);
      acc0[nt] = __builtin_amdgcn_wmma_f32_16x16x32_f16(
          false, af0, false, bfrag, (short)0, acc0[nt], false, false);
      acc1[nt] = __builtin_amdgcn_wmma_f32_16x16x32_f16(
          false, af1, false, bfrag, (short)0, acc1[nt], false, false);
    }
    __syncthreads();
  }

  // epilogue: D layout: lane -> col n = nt*16+l16; VGPR r -> row rbase + r + 8*hi
#pragma unroll
  for (int nt = 0; nt < 8; ++nt) {
    int n = nt * 16 + l16;
    float bv = bias ? bias[n] : 0.0f;
#pragma unroll
    for (int r = 0; r < 8; ++r) {
      int rowa = r0 + r + 8 * hi;
      if (rowa < M) {
        float x = acc0[nt][r] + bv;
        if (SILU) x = siluf(x);
        if (resid) x += resid[(size_t)rowa * 128 + n];
        if (outF) outF[(size_t)rowa * 128 + n] = x;
        if (outH) outH[aswz(rowa, n, 4)] = (_Float16)x;
      }
      int rowb = r1 + r + 8 * hi;
      if (rowb < M) {
        float x = acc1[nt][r] + bv;
        if (SILU) x = siluf(x);
        if (resid) x += resid[(size_t)rowb * 128 + n];
        if (outF) outF[(size_t)rowb * 128 + n] = x;
        if (outH) outH[aswz(rowb, n, 4)] = (_Float16)x;
      }
    }
  }
}

// ---------------------------------------------------------------- elementwise / gather
__global__ void k_radial(const float* __restrict__ coord, const int* __restrict__ ei,
                         float* __restrict__ cd, float* __restrict__ radial){
  int e = blockIdx.x*blockDim.x + threadIdx.x;
  if (e >= Ne) return;
  int r = ei[e], c = ei[Ne + e];
  float s = 0.f;
#pragma unroll
  for (int j = 0; j < 3; ++j){
    float d = coord[r*3+j] - coord[c*3+j];
    cd[(long)e*3+j] = d; s += d*d;
  }
  radial[e] = s;
}

__global__ void k_gather_edge(const float* __restrict__ nf, const float* __restrict__ eattr,
                              const float* __restrict__ radial, const int* __restrict__ ei,
                              _Float16* __restrict__ Ae){
  long g = (long)blockIdx.x*blockDim.x + threadIdx.x;
  if (g >= (long)Ne*72) return;
  int e  = (int)(g / 72);
  int c0 = (int)(g % 72) * 4;
  int r = ei[e], c = ei[Ne + e];
#pragma unroll
  for (int j = 0; j < 4; ++j){
    int d = c0 + j; float v;
    if (d < 128)       v = nf[(long)r*128 + d];
    else if (d < 256)  v = nf[(long)c*128 + (d-128)];
    else if (d == 256) v = radial[e];
    else if (d < 261)  v = eattr[(long)e*4 + (d-257)];
    else               v = 0.0f;
    Ae[aswz(e, d, 9)] = (_Float16)v;
  }
}

__global__ void k_batch_stats(const float* __restrict__ coord, const int* __restrict__ batch,
                              float* __restrict__ csum, float* __restrict__ cntb){
  int i = blockIdx.x*blockDim.x + threadIdx.x;
  if (i >= Nn) return;
  int b = batch[i];
  atomicAdd(&cntb[b], 1.0f);
#pragma unroll
  for (int j = 0; j < 3; ++j) atomicAdd(&csum[b*3+j], coord[i*3+j]);
}

__global__ void k_mx(const float* __restrict__ vc, const float* __restrict__ csum,
                     const float* __restrict__ cntb, float* __restrict__ mX){
  int t = threadIdx.x;
  if (t >= Bb*16) return;
  int b = t >> 4, c = (t >> 2) & 3, k = t & 3;
  float inv = 1.0f / fmaxf(cntb[b], 1.0f);
  float s = 0.f;
#pragma unroll
  for (int j = 0; j < 3; ++j){
    float m = csum[b*3+j] * inv;
    s += (vc[(b*3+j)*4 + c] - m) * (vc[(b*3+j)*4 + k] - m);
  }
  mX[b*16 + c*4 + k] = s;
}

__global__ void k_vcd(const float* __restrict__ coord, const float* __restrict__ vc,
                      const int* __restrict__ batch, float* __restrict__ vcd,
                      float* __restrict__ vrad){
  int g = blockIdx.x*blockDim.x + threadIdx.x;
  if (g >= Nn*Cc) return;
  int i = g >> 2, c = g & 3, b = batch[i];
  float s = 0.f;
#pragma unroll
  for (int j = 0; j < 3; ++j){
    float d = vc[(b*3+j)*4 + c] - coord[i*3+j];
    vcd[((long)i*3+j)*4 + c] = d; s += d*d;
  }
  vrad[g] = sqrtf(s);
}

__global__ void k_gather_virt(const float* __restrict__ nf, const float* __restrict__ vnf,
                              const float* __restrict__ vrad, const float* __restrict__ mX,
                              const int* __restrict__ batch, _Float16* __restrict__ Av){
  long g = (long)blockIdx.x*blockDim.x + threadIdx.x;
  if (g >= (long)Mv*72) return;
  int rowid = (int)(g / 72);
  int i = rowid >> 2, c = rowid & 3;
  int b = batch[i];
  int c0 = (int)(g % 72) * 4;
#pragma unroll
  for (int j = 0; j < 4; ++j){
    int d = c0 + j; float v;
    if (d < 128)       v = nf[(long)i*128 + d];
    else if (d < 256)  v = vnf[((long)b*128 + (d-128))*4 + c];
    else if (d == 256) v = vrad[rowid];
    else if (d < 261)  v = mX[b*16 + (d-257)*4 + c];
    else               v = 0.0f;
    Av[aswz(rowid, d, 9)] = (_Float16)v;
  }
}

__global__ void k_nf_h(const float* __restrict__ nf, _Float16* __restrict__ nfh){
  long g = (long)blockIdx.x*blockDim.x + threadIdx.x;
  if (g >= (long)Nn*128) return;
  int i = (int)(g >> 7), d = (int)(g & 127);
  nfh[aswz(i, d, 4)] = (_Float16)nf[g];
}

// x (f32, post-silu) -> gated in place + swizzled f16 copy; one wave32 per row
__global__ __launch_bounds__(256) void k_attn_gate(float* __restrict__ x, _Float16* __restrict__ xh,
    const float* __restrict__ w, const float* __restrict__ b, int M){
  int wave = threadIdx.x >> 5, lane = threadIdx.x & 31;
  int row = blockIdx.x*8 + wave;
  if (row >= M) return;
  float4 xv = ((const float4*)(x + (long)row*128))[lane];
  float4 wv = ((const float4*)w)[lane];
  float s = xv.x*wv.x + xv.y*wv.y + xv.z*wv.z + xv.w*wv.w;
  for (int off = 16; off > 0; off >>= 1) s += __shfl_down(s, off, 32);
  s = __shfl(s, 0, 32);
  float g = sigmf(s + b[0]);
  xv.x *= g; xv.y *= g; xv.z *= g; xv.w *= g;
  ((float4*)(x + (long)row*128))[lane] = xv;
  v4h hv = { (_Float16)xv.x, (_Float16)xv.y, (_Float16)xv.z, (_Float16)xv.w };
  *(v4h*)(xh + aswz(row, lane*4, 4)) = hv;   // 4 cols stay contiguous under swizzle
}

// out[row] = dot(x_f16_swizzled[row,:128], w) + (bias?bias[0]:0); one wave32 per row
__global__ __launch_bounds__(256) void k_rowdot(const _Float16* __restrict__ x,
    const float* __restrict__ w, const float* __restrict__ bias,
    float* __restrict__ out, int M){
  int wave = threadIdx.x >> 5, lane = threadIdx.x & 31;
  int row = blockIdx.x*8 + wave;
  if (row >= M) return;
  v4h hv = *(const v4h*)(x + aswz(row, lane*4, 4));
  float4 wv = ((const float4*)w)[lane];
  float s = (float)hv[0]*wv.x + (float)hv[1]*wv.y + (float)hv[2]*wv.z + (float)hv[3]*wv.w;
  for (int off = 16; off > 0; off >>= 1) s += __shfl_down(s, off, 32);
  if (lane == 0) out[row] = s + (bias ? bias[0] : 0.0f);
}

__global__ void k_scatter_coord(const int* __restrict__ ei, const float* __restrict__ cd,
                                const float* __restrict__ phir, float* __restrict__ cacc,
                                float* __restrict__ cntr){
  int e = blockIdx.x*blockDim.x + threadIdx.x;
  if (e >= Ne) return;
  int r = ei[e]; float p = phir[e];
  atomicAdd(&cntr[r], 1.0f);
#pragma unroll
  for (int j = 0; j < 3; ++j) atomicAdd(&cacc[r*3+j], cd[(long)e*3+j] * p);
}

__global__ void k_scatter_hacc(const int* __restrict__ ei, const float* __restrict__ ef,
                               float* __restrict__ hacc){
  long g = (long)blockIdx.x*blockDim.x + threadIdx.x;
  if (g >= (long)Ne*32) return;
  int e = (int)(g >> 5);
  int c0 = (int)(g & 31) * 4;
  int r = ei[e];
#pragma unroll
  for (int j = 0; j < 4; ++j)
    atomicAdd(&hacc[(long)r*128 + c0 + j], ef[(long)e*128 + c0 + j]);
}

__global__ void k_scatter_vacc(const int* __restrict__ batch, const float* __restrict__ vcd,
                               const float* __restrict__ phivv, float* __restrict__ vacc){
  int i = blockIdx.x*blockDim.x + threadIdx.x;
  if (i >= Nn) return;
  int b = batch[i];
#pragma unroll
  for (int j = 0; j < 3; ++j)
#pragma unroll
    for (int c = 0; c < 4; ++c)
      atomicAdd(&vacc[b*12 + j*4 + c], vcd[((long)i*3+j)*4+c] * phivv[i*4+c]);
}

__global__ void k_scatter_vnacc(const int* __restrict__ batch, const float* __restrict__ vef,
                                float* __restrict__ vnacc){
  long g = (long)blockIdx.x*blockDim.x + threadIdx.x;
  if (g >= (long)Mv*32) return;
  int rid = (int)(g >> 5);
  int hb  = (int)(g & 31) * 4;
  int b = batch[rid >> 2], c = rid & 3;
#pragma unroll
  for (int j = 0; j < 4; ++j)
    atomicAdd(&vnacc[b*512 + (hb+j)*4 + c], vef[(long)rid*128 + hb + j]);
}

__global__ void k_coord_out(const float* __restrict__ coord, const float* __restrict__ nvel,
                            const float* __restrict__ cacc, const float* __restrict__ cntr,
                            const float* __restrict__ vcd, const float* __restrict__ phirv,
                            const float* __restrict__ phivel, float* __restrict__ outc){
  int i = blockIdx.x*blockDim.x + threadIdx.x;
  if (i >= Nn) return;
  float inv = 1.0f / fmaxf(cntr[i], 1.0f);
  float pv = phivel[i];
#pragma unroll
  for (int j = 0; j < 3; ++j){
    float t2 = 0.f;
#pragma unroll
    for (int c = 0; c < 4; ++c) t2 += vcd[((long)i*3+j)*4+c] * phirv[i*4+c];
    outc[(long)i*3+j] = coord[i*3+j] + cacc[i*3+j]*inv - 0.25f*t2 + pv*nvel[i*3+j];
  }
}

__global__ void k_vc_out(const float* __restrict__ vc, const float* __restrict__ vacc,
                         const float* __restrict__ cntb, float* __restrict__ outvc){
  int idx = blockIdx.x*blockDim.x + threadIdx.x;
  if (idx >= Bb*12) return;
  outvc[idx] = vc[idx] + vacc[idx] / fmaxf(cntb[idx/12], 1.0f);
}

__global__ void k_gather_node(const float* __restrict__ nf, const float* __restrict__ hacc,
                              const float* __restrict__ cntr, const float* __restrict__ vef,
                              const float* __restrict__ nattr, _Float16* __restrict__ An){
  long g = (long)blockIdx.x*blockDim.x + threadIdx.x;
  if (g >= (long)Nn*104) return;
  int i  = (int)(g / 104);
  int c0 = (int)(g % 104) * 4;
  float inv = 1.0f / fmaxf(cntr[i], 1.0f);
#pragma unroll
  for (int j = 0; j < 4; ++j){
    int d = c0 + j; float v;
    if (d < 128)       v = nf[(long)i*128 + d];
    else if (d < 256)  v = hacc[(long)i*128 + (d-128)] * inv;
    else if (d < 384){
      int h = d - 256;
      v = 0.25f * (vef[((long)i*4+0)*128+h] + vef[((long)i*4+1)*128+h]
                 + vef[((long)i*4+2)*128+h] + vef[((long)i*4+3)*128+h]);
    }
    else if (d < 392)  v = nattr[(long)i*8 + (d-384)];
    else               v = 0.0f;
    An[aswz(i, d, 13)] = (_Float16)v;
  }
}

__global__ void k_gather_vin(const float* __restrict__ vnf, const float* __restrict__ vnacc,
                             const float* __restrict__ cntb, _Float16* __restrict__ Avn){
  long g = (long)blockIdx.x*blockDim.x + threadIdx.x;
  if (g >= (long)64*64) return;
  int rowid = (int)(g / 64);
  int b = rowid >> 2, c = rowid & 3;
  int c0 = (int)(g % 64) * 4;
  float inv = 1.0f / fmaxf(cntb[b], 1.0f);
#pragma unroll
  for (int j = 0; j < 4; ++j){
    int d = c0 + j; float v;
    if (d < 128) v = vnf[((long)b*128 + d)*4 + c];
    else         v = vnacc[b*512 + (d-128)*4 + c] * inv;
    Avn[aswz(rowid, d, 8)] = (_Float16)v;
  }
}

__global__ void k_vnf_out(const float* __restrict__ vnf, const float* __restrict__ voutf,
                          float* __restrict__ out){
  int idx = blockIdx.x*blockDim.x + threadIdx.x;
  if (idx >= Bb*128*4) return;
  int b = idx >> 9, f = (idx >> 2) & 127, c = idx & 3;
  out[idx] = vnf[idx] + voutf[((long)(b*4+c))*128 + f];
}

// ---------------------------------------------------------------- launch
extern "C" void kernel_launch(void* const* d_in, const int* in_sizes, int n_in,
                              void* d_out, int out_size, void* d_ws, size_t ws_size,
                              hipStream_t stream) {
  (void)in_sizes; (void)n_in; (void)out_size; (void)ws_size;

  // ---- inputs (setup_inputs dict order; params dict flattened in literal order)
  const float* node_feat = (const float*)d_in[0];
  const float* coord     = (const float*)d_in[1];
  const float* node_vel  = (const float*)d_in[2];
  const float* vc_in     = (const float*)d_in[3];   // [B,3,C]
  const float* vnf_in    = (const float*)d_in[4];   // [B,128,C]
  const float* eattr     = (const float*)d_in[5];
  const float* nattr     = (const float*)d_in[6];
  const float *We1=(const float*)d_in[7],  *be1=(const float*)d_in[8];
  const float *We2=(const float*)d_in[9],  *be2=(const float*)d_in[10];
  const float *Wev1=(const float*)d_in[11],*bev1=(const float*)d_in[12];
  const float *Wev2=(const float*)d_in[13],*bev2=(const float*)d_in[14];
  const float *Wa=(const float*)d_in[15],  *ba=(const float*)d_in[16];
  const float *Wav=(const float*)d_in[17], *bav=(const float*)d_in[18];
  const float *Wr1=(const float*)d_in[19], *br1=(const float*)d_in[20];
  const float *Wr2=(const float*)d_in[21];
  const float *Wrv1=(const float*)d_in[22],*brv1=(const float*)d_in[23];
  const float *Wrv2=(const float*)d_in[24];
  const float *Wvv1=(const float*)d_in[25],*bvv1=(const float*)d_in[26];
  const float *Wvv2=(const float*)d_in[27];
  const float *Wvel1=(const float*)d_in[28],*bvel1=(const float*)d_in[29];
  const float *Wvel2=(const float*)d_in[30],*bvel2=(const float*)d_in[31];
  const float *Wn1=(const float*)d_in[32], *bn1=(const float*)d_in[33];
  const float *Wn2=(const float*)d_in[34], *bn2=(const float*)d_in[35];
  const float *Wnv1=(const float*)d_in[36],*bnv1=(const float*)d_in[37];
  const float *Wnv2=(const float*)d_in[38],*bnv2=(const float*)d_in[39];
  const int* ei    = (const int*)d_in[40];          // [2,E]
  const int* batch = (const int*)d_in[41];          // [N]

  // ---- workspace bump allocator (256B aligned slots)
  char* base = (char*)d_ws;
  size_t off = 0;
  auto alloc = [&](size_t bytes) -> void* {
    void* p = base + off;
    off = (off + bytes + 255) & ~(size_t)255;
    return p;
  };
  // accumulators (zeroed every call; contiguous region)
  float* csum  = (float*)alloc((size_t)Bb*3*4);
  float* cntb  = (float*)alloc((size_t)Bb*4);
  float* cacc  = (float*)alloc((size_t)Nn*3*4);
  float* cntr  = (float*)alloc((size_t)Nn*4);
  float* hacc  = (float*)alloc((size_t)Nn*128*4);
  float* vacc  = (float*)alloc((size_t)Bb*12*4);
  float* vnacc = (float*)alloc((size_t)Bb*512*4);
  size_t zbytes = off;
  // non-zeroed scratch
  float* mX    = (float*)alloc((size_t)Bb*16*4);
  float* vcd   = (float*)alloc((size_t)Nn*12*4);
  float* vrad  = (float*)alloc((size_t)Nn*4*4);
  float* cd    = (float*)alloc((size_t)Ne*3*4);
  float* radial= (float*)alloc((size_t)Ne*4);
  float* phir  = (float*)alloc((size_t)Ne*4);
  float* phirv = (float*)alloc((size_t)Mv*4);
  float* phivv = (float*)alloc((size_t)Mv*4);
  float* phivel= (float*)alloc((size_t)Nn*4);
  _Float16* Ae   = (_Float16*)alloc((size_t)Ne*288*2);
  _Float16* Th   = (_Float16*)alloc((size_t)Ne*128*2);   // reused scratch f16 [Ne,128]
  float*    efF  = (float*)alloc((size_t)Ne*128*4);
  _Float16* efH  = (_Float16*)alloc((size_t)Ne*128*2);
  _Float16* Av   = (_Float16*)alloc((size_t)Mv*288*2);
  _Float16* hv1  = (_Float16*)alloc((size_t)Mv*128*2);   // reused scratch f16 [Mv,128]
  float*    vefF = (float*)alloc((size_t)Mv*128*4);
  _Float16* vefH = (_Float16*)alloc((size_t)Mv*128*2);
  _Float16* nfH  = (_Float16*)alloc((size_t)Nn*128*2);
  _Float16* An   = (_Float16*)alloc((size_t)Nn*416*2);
  _Float16* Avn  = (_Float16*)alloc((size_t)64*256*2);
  _Float16* Tvn  = (_Float16*)alloc((size_t)64*128*2);
  float*    voutF= (float*)alloc((size_t)64*128*4);
  // f16 transposed weights
  _Float16* We1t  = (_Float16*)alloc((size_t)128*288*2);
  _Float16* We2t  = (_Float16*)alloc((size_t)128*128*2);
  _Float16* Wev1t = (_Float16*)alloc((size_t)128*288*2);
  _Float16* Wev2t = (_Float16*)alloc((size_t)128*128*2);
  _Float16* Wr1t  = (_Float16*)alloc((size_t)128*128*2);
  _Float16* Wrv1t = (_Float16*)alloc((size_t)128*128*2);
  _Float16* Wvv1t = (_Float16*)alloc((size_t)128*128*2);
  _Float16* Wvel1t= (_Float16*)alloc((size_t)128*128*2);
  _Float16* Wn1t  = (_Float16*)alloc((size_t)128*416*2);
  _Float16* Wn2t  = (_Float16*)alloc((size_t)128*128*2);
  _Float16* Wnv1t = (_Float16*)alloc((size_t)128*256*2);
  _Float16* Wnv2t = (_Float16*)alloc((size_t)128*128*2);

  // ---- output regions (concatenated flat, f32)
  float* out_nf    = (float*)d_out;                    // [N,128]
  float* out_coord = out_nf + (size_t)Nn*128;          // [N,3]
  float* out_vc    = out_coord + (size_t)Nn*3;         // [B,3,C]
  float* out_vnf   = out_vc + (size_t)Bb*12;           // [B,128,C]

  dim3 blk(256);
  auto nb = [](long n){ return dim3((unsigned)((n + 255) / 256)); };
  auto gb = [](int M){ return dim3((unsigned)((M + 255) / 256)); };
  auto rb = [](int M){ return dim3((unsigned)((M + 7) / 8)); };

  // 0) zero accumulators + weight prep
  k_zero<<<nb((long)(zbytes/4)), blk, 0, stream>>>((float*)base, (long)(zbytes/4));
  k_wt<<<nb(128L*288), blk, 0, stream>>>(We1,  We1t,  261, 288);
  k_wt<<<nb(128L*128), blk, 0, stream>>>(We2,  We2t,  128, 128);
  k_wt<<<nb(128L*288), blk, 0, stream>>>(Wev1, Wev1t, 261, 288);
  k_wt<<<nb(128L*128), blk, 0, stream>>>(Wev2, Wev2t, 128, 128);
  k_wt<<<nb(128L*128), blk, 0, stream>>>(Wr1,  Wr1t,  128, 128);
  k_wt<<<nb(128L*128), blk, 0, stream>>>(Wrv1, Wrv1t, 128, 128);
  k_wt<<<nb(128L*128), blk, 0, stream>>>(Wvv1, Wvv1t, 128, 128);
  k_wt<<<nb(128L*128), blk, 0, stream>>>(Wvel1,Wvel1t,128, 128);
  k_wt<<<nb(128L*416), blk, 0, stream>>>(Wn1,  Wn1t,  392, 416);
  k_wt<<<nb(128L*128), blk, 0, stream>>>(Wn2,  Wn2t,  128, 128);
  k_wt<<<nb(128L*256), blk, 0, stream>>>(Wnv1, Wnv1t, 256, 256);
  k_wt<<<nb(128L*128), blk, 0, stream>>>(Wnv2, Wnv2t, 128, 128);

  // 1) edge path
  k_radial<<<nb(Ne), blk, 0, stream>>>(coord, ei, cd, radial);
  k_gather_edge<<<nb((long)Ne*72), blk, 0, stream>>>(node_feat, eattr, radial, ei, Ae);
  gemm128<288,true><<<gb(Ne), blk, 0, stream>>>(Ae, We1t, be1, nullptr, nullptr, Th, Ne);
  gemm128<128,true><<<gb(Ne), blk, 0, stream>>>(Th, We2t, be2, nullptr, efF, nullptr, Ne);
  k_attn_gate<<<rb(Ne), blk, 0, stream>>>(efF, efH, Wa, ba, Ne);

  // 2) virtual edge path
  k_batch_stats<<<nb(Nn), blk, 0, stream>>>(coord, batch, csum, cntb);
  k_mx<<<dim3(1), blk, 0, stream>>>(vc_in, csum, cntb, mX);
  k_vcd<<<nb(Nn*Cc), blk, 0, stream>>>(coord, vc_in, batch, vcd, vrad);
  k_gather_virt<<<nb((long)Mv*72), blk, 0, stream>>>(node_feat, vnf_in, vrad, mX, batch, Av);
  gemm128<288,true><<<gb(Mv), blk, 0, stream>>>(Av, Wev1t, bev1, nullptr, nullptr, hv1, Mv);
  gemm128<128,true><<<gb(Mv), blk, 0, stream>>>(hv1, Wev2t, bev2, nullptr, vefF, nullptr, Mv);
  k_attn_gate<<<rb(Mv), blk, 0, stream>>>(vefF, vefH, Wav, bav, Mv);

  // 3) coordinate heads
  gemm128<128,true><<<gb(Ne), blk, 0, stream>>>(efH, Wr1t, br1, nullptr, nullptr, Th, Ne);
  k_rowdot<<<rb(Ne), blk, 0, stream>>>(Th, Wr2, nullptr, phir, Ne);
  gemm128<128,true><<<gb(Mv), blk, 0, stream>>>(vefH, Wrv1t, brv1, nullptr, nullptr, hv1, Mv);
  k_rowdot<<<rb(Mv), blk, 0, stream>>>(hv1, Wrv2, nullptr, phirv, Mv);
  gemm128<128,true><<<gb(Mv), blk, 0, stream>>>(vefH, Wvv1t, bvv1, nullptr, nullptr, hv1, Mv);
  k_rowdot<<<rb(Mv), blk, 0, stream>>>(hv1, Wvv2, nullptr, phivv, Mv);
  k_nf_h<<<nb((long)Nn*128), blk, 0, stream>>>(node_feat, nfH);
  gemm128<128,true><<<gb(Nn), blk, 0, stream>>>(nfH, Wvel1t, bvel1, nullptr, nullptr, Th, Nn);
  k_rowdot<<<rb(Nn), blk, 0, stream>>>(Th, Wvel2, bvel2, phivel, Nn);

  // 4) segment reductions
  k_scatter_coord<<<nb(Ne), blk, 0, stream>>>(ei, cd, phir, cacc, cntr);
  k_scatter_hacc<<<nb((long)Ne*32), blk, 0, stream>>>(ei, efF, hacc);
  k_scatter_vacc<<<nb(Nn), blk, 0, stream>>>(batch, vcd, phivv, vacc);
  k_scatter_vnacc<<<nb((long)Mv*32), blk, 0, stream>>>(batch, vefF, vnacc);

  // 5) coord & virtual-coord outputs
  k_coord_out<<<nb(Nn), blk, 0, stream>>>(coord, node_vel, cacc, cntr, vcd, phirv, phivel, out_coord);
  k_vc_out<<<nb(Bb*12), blk, 0, stream>>>(vc_in, vacc, cntb, out_vc);

  // 6) node model (residual fused into GEMM epilogue, writes d_out directly)
  k_gather_node<<<nb((long)Nn*104), blk, 0, stream>>>(node_feat, hacc, cntr, vefF, nattr, An);
  gemm128<416,true><<<gb(Nn), blk, 0, stream>>>(An, Wn1t, bn1, nullptr, nullptr, Th, Nn);
  gemm128<128,false><<<gb(Nn), blk, 0, stream>>>(Th, Wn2t, bn2, node_feat, out_nf, nullptr, Nn);

  // 7) virtual node model
  k_gather_vin<<<nb(64L*64), blk, 0, stream>>>(vnf_in, vnacc, cntb, Avn);
  gemm128<256,true><<<gb(64), blk, 0, stream>>>(Avn, Wnv1t, bnv1, nullptr, nullptr, Tvn, 64);
  gemm128<128,false><<<gb(64), blk, 0, stream>>>(Tvn, Wnv2t, bnv2, nullptr, voutF, nullptr, 64);
  k_vnf_out<<<nb(Bb*128L*4), blk, 0, stream>>>(vnf_in, voutF, out_vnf);
}
